// GRU_71347996721191
// MI455X (gfx1250) — matrix operand order repaired and checked by reading, not compiled, not run on hardware
//
#include <hip/hip_runtime.h>

#define DD 256   // input dim
#define HH 256   // hidden dim
#define G3 768   // 3*H

typedef __bf16 v16bf __attribute__((ext_vector_type(16)));
typedef float  v8f   __attribute__((ext_vector_type(8)));

union TileU { v16bf v; uint4 q[2]; };
union TileC { v8f v; float f[8]; };

static __device__ __forceinline__ unsigned short f2bf(float f) {
    unsigned int u = __float_as_uint(f);
    u += 0x7FFFu + ((u >> 16) & 1u);          // round-to-nearest-even
    return (unsigned short)(u >> 16);
}

// A operand: 16x32 bf16 tile, row-major src [*, ld].
// ISA layout: lanes 0-15 = M rows, K 0-7 & 16-23; lanes 16-31 same rows, K 8-15 & 24-31.
static __device__ __forceinline__ v16bf load_tileA(const unsigned short* src, int ld,
                                                   int rbase, int kb) {
    int lane = threadIdx.x & 31;
    int r  = rbase + (lane & 15);
    int hi = lane >> 4;
    const unsigned short* p = src + (size_t)r * ld + kb + hi * 8;
    TileU t;
    t.q[0] = *(const uint4*)(p);       // k: kb + hi*8 + 0..7
    t.q[1] = *(const uint4*)(p + 16);  // k: kb + 16 + hi*8 + 0..7
    return t.v;
}

// B operand: 32x16 bf16 tile where B[k][n] = src[n][k] (src row-major [N, ld]).
// ISA layout: lanes 0-15 = column n, K 0-7 & 16-23; lanes 16-31 = column n, K 8-15 & 24-31.
static __device__ __forceinline__ v16bf load_tileB(const unsigned short* src, int ld,
                                                   int nbase, int kb) {
    int lane = threadIdx.x & 31;
    int n  = nbase + (lane & 15);
    int hi = lane >> 4;
    const unsigned short* p = src + (size_t)n * ld + kb + hi * 8;
    TileU t;
    t.q[0] = *(const uint4*)(p);       // k: kb + hi*8 + 0..7
    t.q[1] = *(const uint4*)(p + 16);  // k: kb + 16 + hi*8 + 0..7
    return t.v;
}

__global__ void cvt_bf16(const float* __restrict__ src, unsigned short* __restrict__ dst, int n) {
    int i = blockIdx.x * blockDim.x + threadIdx.x;
    if (i < n) dst[i] = f2bf(src[i]);
}

// -------- Phase 1: igates[T,768] = x @ W_ih^T + bias, WMMA bf16 ----------
__global__ __launch_bounds__(256) void igates_gemm(
    const unsigned short* __restrict__ xb,    // [T][256] bf16
    const unsigned short* __restrict__ wihb,  // [768][256] bf16
    const float* __restrict__ bias,           // [768]
    float* __restrict__ ig,                   // [T][768]
    int mtiles)
{
    int wv = (blockIdx.x * blockDim.x + threadIdx.x) >> 5;  // global wave id
    int mtile = wv / 48;
    int ntile = wv % 48;
    if (mtile >= mtiles) return;
    int lane = threadIdx.x & 31;
    int hi  = lane >> 4;
    int col = lane & 15;

    v8f c = {};
    for (int kt = 0; kt < 8; ++kt) {
        v16bf a = load_tileA(xb,   DD, mtile * 16, kt * 32);
        v16bf b = load_tileB(wihb, DD, ntile * 16, kt * 32);
        c = __builtin_amdgcn_wmma_f32_16x16x32_bf16(false, a, false, b,
                                                    (short)0, c, false, false);
    }
    int g = ntile * 16 + col;
    float bv = bias[g];
    TileC tc; tc.v = c;
    for (int v = 0; v < 8; ++v) {
        int t = mtile * 16 + v + hi * 8;
        ig[(size_t)t * G3 + g] = tc.f[v] + bv;
    }
}

// -------- Segment extraction: start-resets make segments independent ----------
__global__ void find_segs(const int* __restrict__ start, int T,
                          int* __restrict__ segb, int* __restrict__ cnt)
{
    int t = blockIdx.x * blockDim.x + threadIdx.x;
    if (t >= T) return;
    if (t == 0 || start[t] != 0) {
        int i = atomicAdd(cnt, 1);
        segb[i] = t;
    }
}

// -------- Phase 2: batched segment scan, 16 segments = WMMA N dimension ----------
__global__ __launch_bounds__(256) void gru_scan(
    const float* __restrict__ ig,            // [T][768]
    const unsigned short* __restrict__ whh,  // [768][256] bf16 (L2-resident)
    const float* __restrict__ bias_n,        // [256]
    const float* __restrict__ state0,        // [256]
    const int* __restrict__ start,           // [T]
    const int* __restrict__ segb,
    const int* __restrict__ segcnt,
    float* __restrict__ out,                 // y[T][256] ++ final_state[256]
    int T)
{
    __shared__ int sB[16], sL[16], sMax;
    __shared__ __align__(16) unsigned short hB[16 * 256]; // bf16 h, [seg][k] (B operand)
    __shared__ float hF[16 * 256];                        // f32 h master, [seg][k]
    __shared__ float hg[768 * 16];                        // W_hh @ h, [row][seg]

    int tid = threadIdx.x;
    int cnt = *segcnt;
    if (blockIdx.x * 16 >= cnt) return;   // block-uniform early out

    if (tid < 16) {
        int s = blockIdx.x * 16 + tid;
        int b = 0, L = 0;
        if (s < cnt) {
            b = segb[s];
            int e = b + 1;
            while (e < T && start[e] == 0) e++;   // geometric, E[len]~2
            L = e - b;
        }
        sB[tid] = b; sL[tid] = L;
    }
    __syncthreads();
    if (tid == 0) {
        int m = 0;
        for (int j = 0; j < 16; ++j) m = max(m, sL[j]);
        sMax = m;
    }
    // initial h per segment: the t==0 segment starts from `state` unless start[0] resets
    for (int j = 0; j < 16; ++j) {
        float h0 = 0.f;
        if (sL[j] > 0 && sB[j] == 0 && start[0] == 0) h0 = state0[tid];
        hF[j * 256 + tid] = h0;
        hB[j * 256 + tid] = f2bf(h0);
    }
    __syncthreads();

    int maxLen = sMax;
    int lane = tid & 31;
    int wv   = tid >> 5;     // 8 waves
    int hi   = lane >> 4;
    int col  = lane & 15;

    for (int step = 0; step < maxLen; ++step) {
        // B tiles: h columns of all 16 segments, shared across this wave's row tiles
        v16bf b[8];
        for (int kt = 0; kt < 8; ++kt)
            b[kt] = load_tileB(hB, HH, 0, kt * 32);

        // each wave owns 6 of the 48 output row tiles (rows of W_hh)
        for (int i = 0; i < 6; ++i) {
            int rt = wv * 6 + i;
            v8f c = {};
            for (int kt = 0; kt < 8; ++kt) {
                v16bf a = load_tileA(whh, HH, rt * 16, kt * 32);
                c = __builtin_amdgcn_wmma_f32_16x16x32_bf16(false, a, false, b[kt],
                                                            (short)0, c, false, false);
            }
            TileC tc; tc.v = c;
            for (int v = 0; v < 8; ++v) {
                int row = rt * 16 + v + hi * 8;
                hg[row * 16 + col] = tc.f[v];
            }
        }
        __syncthreads();

        // gate math: iteration j handles segment j with f = tid fully coalesced
        for (int j = 0; j < 16; ++j) {
            bool active = (step < sL[j]);
            int f = tid;
            size_t tg = active ? (size_t)(sB[j] + step) : 0;
            float igr = ig[tg * G3 + f];
            float igz = ig[tg * G3 + 256 + f];
            float ign = ig[tg * G3 + 512 + f];
            float hgr = hg[f * 16 + j];
            float hgz = hg[(256 + f) * 16 + j];
            float hgn = hg[(512 + f) * 16 + j];
            float h   = hF[j * 256 + f];
            float r = 1.f / (1.f + __expf(-(igr + hgr)));
            float z = 1.f / (1.f + __expf(-(igz + hgz)));
            float n = tanhf(ign + r * (hgn + bias_n[f]));
            float hn = n + z * (h - n);
            if (active) {
                out[tg * HH + f] = hn;
                if ((int)tg == T - 1) out[(size_t)T * HH + f] = hn;  // final_state
                hF[j * 256 + f] = hn;
                hB[j * 256 + f] = f2bf(hn);
            }
        }
        __syncthreads();
    }
}

extern "C" void kernel_launch(void* const* d_in, const int* in_sizes, int n_in,
                              void* d_out, int out_size, void* d_ws, size_t ws_size,
                              hipStream_t stream)
{
    (void)n_in; (void)out_size; (void)ws_size;
    const float* x      = (const float*)d_in[0];
    const float* state  = (const float*)d_in[1];
    const int*   start  = (const int*)d_in[2];
    /* d_in[3] = next_done, unused by forward */
    const float* W_ih   = (const float*)d_in[4];
    const float* W_hh   = (const float*)d_in[5];
    const float* bias   = (const float*)d_in[6];
    const float* bias_n = (const float*)d_in[7];
    float* out = (float*)d_out;
    int T = in_sizes[2];

    // workspace carve-out (256B aligned)
    char* ws = (char*)d_ws;
    size_t off = 0;
    auto carve = [&](size_t bytes) -> char* {
        char* p = ws + off;
        off += (bytes + 255) & ~(size_t)255;
        return p;
    };
    unsigned short* xb     = (unsigned short*)carve((size_t)T * DD * 2);
    unsigned short* wihb   = (unsigned short*)carve((size_t)G3 * DD * 2);
    unsigned short* whhb   = (unsigned short*)carve((size_t)G3 * HH * 2);
    float*          igs    = (float*)carve((size_t)T * G3 * 4);
    int*            sb     = (int*)carve((size_t)T * 4);
    int*            scnt   = (int*)carve(256);

    hipMemsetAsync(scnt, 0, sizeof(int), stream);

    int nx = T * DD;
    cvt_bf16<<<(nx + 255) / 256, 256, 0, stream>>>(x, xb, nx);
    cvt_bf16<<<(G3 * DD + 255) / 256, 256, 0, stream>>>(W_ih, wihb, G3 * DD);
    cvt_bf16<<<(G3 * HH + 255) / 256, 256, 0, stream>>>(W_hh, whhb, G3 * HH);

    int mtiles = T / 16;                       // 4096
    int waves  = mtiles * 48;                  // one wave per 16x16 output tile
    igates_gemm<<<waves / 8, 256, 0, stream>>>(xb, wihb, bias, igs, mtiles);

    find_segs<<<(T + 255) / 256, 256, 0, stream>>>(start, T, sb, scnt);

    gru_scan<<<(T + 15) / 16, 256, 0, stream>>>(igs, whhb, bias_n, state, start,
                                                sb, scnt, out, T);
}
